// UserModel_63264868270528
// MI455X (gfx1250) — matrix operand order repaired and checked by reading, not compiled, not run on hardware
//
#include <hip/hip_runtime.h>
#include <hip/hip_bf16.h>

typedef __attribute__((ext_vector_type(16))) _Float16 v16h;
typedef __attribute__((ext_vector_type(8)))  _Float16 v8h;
typedef __attribute__((ext_vector_type(8)))  float    v8f;
typedef __attribute__((ext_vector_type(4)))  float    v4f;
typedef __attribute__((ext_vector_type(4)))  unsigned int u32x4;
typedef __attribute__((ext_vector_type(4)))  int      i32x4;
typedef __attribute__((ext_vector_type(8)))  int      i32x8;

constexpr int N_ = 512, K_ = 32, J_ = 512, L_ = 256, R_ = 4, M_ = 8;
constexpr int NK  = N_ * K_;        // 16384 GEMM rows
constexpr int RMK = R_ * M_ * K_;   // 1024 (r,m,k) combos

// ---------------- K1: base[l,j] = -(d+eps)^0.4 (f32 for Z, f16 for WMMA B) ---
__global__ void base_kernel(const float* __restrict__ dmat,
                            float* __restrict__ base32,
                            _Float16* __restrict__ baseh) {
    int i = blockIdx.x * blockDim.x + threadIdx.x;
    if (i < L_ * J_) {
        float b = -powf(dmat[i] + 1e-8f, 0.4f);
        base32[i] = b;
        baseh[i]  = (_Float16)b;
    }
}

// ---------------- K2: sx[row] = sum_j x[row, j]  (wave32 per row) -----------
__global__ void rowsum_kernel(const float* __restrict__ x, float* __restrict__ sx) {
    int lane = threadIdx.x & 31;
    int row  = blockIdx.x * 8 + (threadIdx.x >> 5);
    const float* p = x + (size_t)row * J_;
    float s = 0.f;
    for (int j = lane; j < J_; j += 32) s += p[j];
    #pragma unroll
    for (int off = 16; off > 0; off >>= 1) s += __shfl_xor(s, off, 32);
    if (lane == 0) sx[row] = s;
}

// ---------------- K3: Z[rmk,l] = logsumexp_j(alpha[rmk,l]*base[l,j]) --------
__global__ void z_kernel(const float* __restrict__ alpha,
                         const float* __restrict__ base32,
                         float* __restrict__ Z) {
    __shared__ float bs[J_];
    int l = blockIdx.x;
    int t = threadIdx.x;
    for (int j = t; j < J_; j += 256) bs[j] = base32[l * J_ + j];
    __syncthreads();
    for (int c = 0; c < RMK / 256; ++c) {
        int combo = c * 256 + t;
        float a  = alpha[(size_t)combo * L_ + l];
        float mx = -3.4e38f;
        for (int j = 0; j < J_; ++j) mx = fmaxf(mx, a * bs[j]);
        float s = 0.f;
        for (int j = 0; j < J_; ++j) s += __expf(a * bs[j] - mx);
        Z[(size_t)combo * L_ + l] = mx + __logf(s);
    }
}

// ---------------- K4: WMMA GEMM  S[nk,l] = sum_j x[nk,j]*base[l,j] ----------
// 8 waves/block, one 16x16 output tile per wave, K-loop 512 in steps of 32.
// B tile (16 rows x 512 halves = 16KB) staged once per block via the Tensor
// Data Mover (TENSOR_LOAD_TO_LDS + s_wait_tensorcnt), shared by all 8 waves.
// Fragment layouts per CDNA5 ISA 7.12.2.
__global__ void gemm_kernel(const float* __restrict__ x,
                            const _Float16* __restrict__ baseh,
                            float* __restrict__ S) {
    __shared__ _Float16 btile[16 * J_];     // 16 KB B tile: rows l0..l0+15
    int wave = threadIdx.x >> 5;
    int lane = threadIdx.x & 31;
    int rt = blockIdx.x * 8 + wave;         // row tile (0..1023)
    int l0 = blockIdx.y * 16;               // col tile base
    int hi = (lane >= 16);
    int mrow = lane & 15;

    // ---- TDM: DMA the B tile into LDS (one descriptor, wave 0 issues) -----
    if (threadIdx.x == 0) {
        unsigned long long ga = (unsigned long long)(uintptr_t)(baseh + (size_t)l0 * J_);
        unsigned lds = (unsigned)(uintptr_t)(&btile[0]);   // flat->LDS offset (addr[31:0])
        // D# group 0: count=1 | lds_addr | global_addr[56:0] | type=2
        u32x4 g0 = { 1u, lds,
                     (unsigned)(ga & 0xFFFFFFFFu),
                     (unsigned)((ga >> 32) & 0x01FFFFFFu) | (2u << 30) };
        // D# group 1: data_size=1(2B); tensor_dim0=J; tensor_dim1=L;
        //             tile_dim0=J; tile_dim1=16; tensor_dim0_stride=J
        i32x8 g1 = { (int)(1u << 16),
                     (int)((unsigned)J_ << 16),
                     (int)((unsigned)L_ << 16),
                     (int)((unsigned)J_ << 16),
                     16,
                     J_,
                     0, 0 };
        i32x4 gz = { 0, 0, 0, 0 };
#if defined(__clang_major__) && __clang_major__ >= 23
        i32x8 gz8 = { 0, 0, 0, 0, 0, 0, 0, 0 };
        __builtin_amdgcn_tensor_load_to_lds(g0, g1, gz, gz, gz8, 0);
#else
        __builtin_amdgcn_tensor_load_to_lds(g0, g1, gz, gz, 0);
#endif
        __builtin_amdgcn_s_wait_tensorcnt(0);
    }
    __syncthreads();

    const float*    xrow = x + (size_t)(rt * 16 + mrow) * J_;
    const _Float16* brow = &btile[(size_t)mrow * J_ + (hi ? 8 : 0)];
    int coff = hi ? 8 : 0;                  // A-fragment K skew for high half-wave

    v8f c = {};
    for (int jt = 0; jt < J_; jt += 32) {
        // A: halves h=0..7 <- K=jt+coff..+7 ; h=8..15 <- K=jt+coff+16..+23
        const v4f* pa0 = (const v4f*)(xrow + jt + coff);
        const v4f* pa1 = (const v4f*)(xrow + jt + coff + 16);
        v4f f0 = pa0[0], f1 = pa0[1], f2 = pa1[0], f3 = pa1[1];
        v16h a;
        #pragma unroll
        for (int i = 0; i < 4; ++i) {
            a[i]      = (_Float16)f0[i];
            a[4 + i]  = (_Float16)f1[i];
            a[8 + i]  = (_Float16)f2[i];
            a[12 + i] = (_Float16)f3[i];
        }
        // B from LDS: halves h=0..7 <- K=jt+coff..+7 ; h=8..15 <- K=jt+coff+16..+23
        const v8h* pb = (const v8h*)(brow + jt);
        v8h b0 = pb[0], b1 = pb[2];
        v16h b;
        #pragma unroll
        for (int i = 0; i < 8; ++i) { b[i] = b0[i]; b[8 + i] = b1[i]; }

        c = __builtin_amdgcn_wmma_f32_16x16x32_f16(false, a, false, b,
                                                   (short)0, c, false, false);
    }
    // C/D layout: VGPR r -> M = r + (hi?8:0), N = lane&15
    int mbase = rt * 16 + (hi ? 8 : 0);
    #pragma unroll
    for (int r = 0; r < 8; ++r)
        S[(size_t)(mbase + r) * L_ + l0 + mrow] = c[r];
}

// ---------------- K5: posterior / entropy (block per n, thread = l) ---------
__device__ inline float blockReduceSum(float v, float* red) {
    int t = threadIdx.x;
    red[t] = v; __syncthreads();
    for (int off = 128; off > 0; off >>= 1) {
        if (t < off) red[t] += red[t + off];
        __syncthreads();
    }
    float r = red[0]; __syncthreads();
    return r;
}
__device__ inline float blockReduceMax(float v, float* red) {
    int t = threadIdx.x;
    red[t] = v; __syncthreads();
    for (int off = 128; off > 0; off >>= 1) {
        if (t < off) red[t] = fmaxf(red[t], red[t + off]);
        __syncthreads();
    }
    float r = red[0]; __syncthreads();
    return r;
}

__global__ void posterior_kernel(const float* __restrict__ S,
                                 const float* __restrict__ sx,
                                 const float* __restrict__ alpha,
                                 const float* __restrict__ Z,
                                 const float* __restrict__ logp,
                                 float* __restrict__ out_ent,
                                 float* __restrict__ out_post) {
    __shared__ float red[256];
    __shared__ float sxs[K_];
    int n = blockIdx.x;
    int l = threadIdx.x;

    float Sreg[K_];
    #pragma unroll
    for (int k = 0; k < K_; ++k) Sreg[k] = S[((size_t)n * K_ + k) * L_ + l];
    if (l < K_) sxs[l] = sx[n * K_ + l];
    __syncthreads();

    float lpv = logp[l];
    float ent_local = 0.f;
    float tp = 0.f;
    for (int r = 0; r < R_; ++r) {
        float mx = -3.4e38f, sm = 0.f;      // streaming logsumexp over m
        for (int m = 0; m < M_; ++m) {
            const float* ap = alpha + (size_t)(r * M_ + m) * K_ * L_ + l;
            const float* zp = Z     + (size_t)(r * M_ + m) * K_ * L_ + l;
            float acc = lpv;
            #pragma unroll
            for (int k = 0; k < K_; ++k)
                acc += ap[k * L_] * Sreg[k] - zp[k * L_] * sxs[k];
            // log_softmax over L across the block
            float mxl   = blockReduceMax(acc, red);
            float e     = __expf(acc - mxl);
            float ssum  = blockReduceSum(e, red);
            float lpost = acc - mxl - __logf(ssum);
            ent_local  -= lpost * __expf(lpost);
            if (lpost > mx) { sm = sm * __expf(mx - lpost) + 1.f; mx = lpost; }
            else            { sm += __expf(lpost - mx); }
        }
        tp += mx + __logf(sm) - __logf((float)M_);
    }
    float ent_tot = blockReduceSum(ent_local, red);
    if (l == 0) out_ent[n] = ent_tot / (float)(M_ * R_);

    // posterior = softmax_l(tp / R)
    float v   = tp / (float)R_;
    float mxp = blockReduceMax(v, red);
    float e   = __expf(v - mxp);
    float sp  = blockReduceSum(e, red);
    out_post[(size_t)n * L_ + l] = e / sp;
}

// ---------------- K6: mean_alpha[k,l] --------------------------------------
__global__ void meanalpha_kernel(const float* __restrict__ alpha,
                                 float* __restrict__ out_mean) {
    int i = blockIdx.x * blockDim.x + threadIdx.x;   // over K_*L_
    if (i < K_ * L_) {
        float s = 0.f;
        for (int rm = 0; rm < R_ * M_; ++rm) s += alpha[(size_t)rm * K_ * L_ + i];
        out_mean[i] = s / (float)(R_ * M_);
    }
}

extern "C" void kernel_launch(void* const* d_in, const int* in_sizes, int n_in,
                              void* d_out, int out_size, void* d_ws, size_t ws_size,
                              hipStream_t stream) {
    const float* x     = (const float*)d_in[0];   // [N,K,J]
    const float* logp  = (const float*)d_in[1];   // [L]
    const float* alpha = (const float*)d_in[2];   // [R,M,K,L]
    const float* dmat  = (const float*)d_in[3];   // [L,J]

    // workspace layout (~18 MB)
    float* base32 = (float*)d_ws;                         // L*J
    float* S      = base32 + L_ * J_;                     // NK*L
    float* sx     = S + (size_t)NK * L_;                  // NK
    float* Z      = sx + NK;                              // RMK*L
    _Float16* baseh = (_Float16*)(Z + (size_t)RMK * L_);  // L*J halves

    float* out_ent  = (float*)d_out;                      // [N]
    float* out_post = out_ent + N_;                       // [N,L]
    float* out_mean = out_post + (size_t)N_ * L_;         // [K,L]

    base_kernel<<<(L_ * J_ + 255) / 256, 256, 0, stream>>>(dmat, base32, baseh);
    rowsum_kernel<<<NK / 8, 256, 0, stream>>>(x, sx);
    z_kernel<<<L_, 256, 0, stream>>>(alpha, base32, Z);
    gemm_kernel<<<dim3(NK / 16 / 8, L_ / 16), 256, 0, stream>>>(x, baseh, S);
    posterior_kernel<<<N_, 256, 0, stream>>>(S, sx, alpha, Z, logp, out_ent, out_post);
    meanalpha_kernel<<<(K_ * L_ + 255) / 256, 256, 0, stream>>>(alpha, out_mean);
}
